// NSE_42640435315484
// MI455X (gfx1250) — matrix operand order repaired and checked by reading, not compiled, not run on hardware
//
#include <hip/hip_runtime.h>
#include <hip/hip_bf16.h>

// NSE scan for MI455X (gfx1250, wave32, WMMA).
// 4 workgroups x 512 threads (16 waves); each workgroup owns 16 batch rows for
// the entire 96-step recurrence. All GEMMs are v_wmma_f32_16x16x32_f16 with
// f16 activations in LDS and f16 weights pre-converted into workspace.

#define S_LEN   96
#define BATCH   64
#define DIM     256
#define BPG     16      // batch rows per workgroup (== WMMA M tile)
#define NTHR    512     // threads per workgroup
#define NW      16      // waves per workgroup (wave32)

typedef __attribute__((ext_vector_type(16))) _Float16 v16h;
typedef __attribute__((ext_vector_type(8)))  _Float16 v8h;
typedef __attribute__((ext_vector_type(8)))  float    v8f;

union V8F { v8f v; float f[8]; };

// A-fragment (16x32 f16, ISA 7.12.2 16-bit A layout):
//  lane<16 : row m=lane,   halves = A[m][k..k+7]  ++ A[m][k+16..k+23]
//  lane>=16: row m=lane-16, same pattern shifted by +8 in K.
__device__ __forceinline__ v16h load_a_frag(const _Float16* base, int ld,
                                            int kbase, int lane) {
  const _Float16* p = base + (lane & 15) * ld + kbase + ((lane >> 4) << 3);
  v8h lo = *(const v8h*)(p);
  v8h hi = *(const v8h*)(p + 16);
  return __builtin_shufflevector(lo, hi, 0,1,2,3,4,5,6,7,8,9,10,11,12,13,14,15);
}

// B-fragment (32x16 f16): lanes 0-15 hold K=kbase..kbase+7 ++ kbase+16..kbase+23 of column n,
// lanes 16-31 same pattern shifted by +8 in K.  wrow = &W[nbase + (lane&15)][0]
// (W row-major [N][K], so B[k][n] = W[n][k]).
__device__ __forceinline__ v16h load_b_frag(const _Float16* wrow, int kbase,
                                            int lane) {
  const _Float16* p = wrow + kbase + ((lane >> 4) << 3);
  v8h lo = *(const v8h*)(p);
  v8h hi = *(const v8h*)(p + 16);
  return __builtin_shufflevector(lo, hi, 0,1,2,3,4,5,6,7,8,9,10,11,12,13,14,15);
}

template<int K>
__device__ __forceinline__ void gemm_acc(v8f& acc, const _Float16* A, int lda,
                                         const _Float16* Wrow, int lane) {
#pragma unroll
  for (int k = 0; k < K; k += 32) {
    v16h a = load_a_frag(A, lda, k, lane);
    v16h b = load_b_frag(Wrow, k, lane);
    acc = __builtin_amdgcn_wmma_f32_16x16x32_f16(
        /*neg_a=*/false, a, /*neg_b=*/false, b,
        /*c_mod=*/(short)0, acc, /*reuse_a=*/false, /*reuse_b=*/false);
  }
}

__device__ __forceinline__ float sigmoidf_fast(float x) {
  return 1.0f / (1.0f + __expf(-x));
}

// ---------------------------------------------------------------- helpers ---

__global__ void NSE_f32_to_f16(const float* __restrict__ src,
                               _Float16* __restrict__ dst, int n) {
  int i = blockIdx.x * blockDim.x + threadIdx.x;
  if (i < n) dst[i] = (_Float16)src[i];
}

// M0[b][s][d] = embed_W[premise[s][b]][d]
__global__ void NSE_init_M(const int* __restrict__ premise,
                           const float* __restrict__ embed_W,
                           float* __restrict__ M) {
  int i = blockIdx.x * blockDim.x + threadIdx.x;
  if (i >= BATCH * S_LEN * DIM) return;
  int d = i & (DIM - 1);
  int s = (i >> 8) % S_LEN;
  int b = i / (S_LEN * DIM);
  M[i] = embed_W[(size_t)premise[s * BATCH + b] * DIM + d];
}

// ------------------------------------------------------------- main scan ---

extern "C" __global__ void __launch_bounds__(NTHR)
NSE_scan(const int* __restrict__ premise, const float* __restrict__ embed_W,
         const _Float16* __restrict__ Wih_r, const _Float16* __restrict__ Whh_r,
         const float* __restrict__ b_r,
         const _Float16* __restrict__ Wx, const float* __restrict__ bx,
         const _Float16* __restrict__ Wih_w, const _Float16* __restrict__ Whh_w,
         const float* __restrict__ b_w,
         float* __restrict__ M) {
  extern __shared__ char smem[];
  float*    gates  = (float*)(smem);                 // [16][1024]  64 KB
  float*    c_r    = (float*)(smem + 65536);         // [16][256]   16 KB
  float*    c_w    = (float*)(smem + 81920);         // [16][256]   16 KB
  float*    zbuf   = (float*)(smem + 98304);         // [16][96]     6 KB
  _Float16* xcomp  = (_Float16*)(smem + 104448);     // x16 [16][256] / comp16 [16][512]  16 KB
  _Float16* cat16  = (_Float16*)(smem + 120832);     // [16][512]: [:,0:256]=h_r, [:,256:512]=m_r
  _Float16* hw16   = (_Float16*)(smem + 137216);     // [16][256]    8 KB
  _Float16* x16    = xcomp;                          // ld = 256
  _Float16* comp16 = xcomp;                          // ld = 512

  const int tid  = threadIdx.x;
  const int lane = tid & 31;
  const int wave = tid >> 5;
  const int b0   = blockIdx.x * BPG;

  // zero recurrent state
  for (int i = tid; i < BPG * DIM; i += NTHR) {
    c_r[i] = 0.0f;
    c_w[i] = 0.0f;
    cat16[(i >> 8) * 512 + (i & 255)] = (_Float16)0.0f;  // h_r(prev) = 0
    hw16[i] = (_Float16)0.0f;                            // h_w(prev) = 0
  }
  __syncthreads();

  for (int t = 0; t < S_LEN; ++t) {
    // ---- embed gather: x16[b][d] = (f16) embed_W[premise[t][b0+b]][d] ----
    for (int i = tid; i < BPG * DIM; i += NTHR) {
      int b = i >> 8, d = i & 255;
      int tok = premise[t * BATCH + b0 + b];
      x16[b * DIM + d] = (_Float16)embed_W[(size_t)tok * DIM + d];
    }
    __syncthreads();

    // ---- GEMM1: gates = x @ Wih_r^T + h_r @ Whh_r^T + b_r  (16x1024) ----
    for (int tt = wave; tt < 64; tt += NW) {
      int n = tt * 16 + (lane & 15);
      float bias = b_r[n];
      V8F acc;
#pragma unroll
      for (int g = 0; g < 8; ++g) acc.f[g] = bias;
      gemm_acc<DIM>(acc.v, x16,   DIM, Wih_r + (size_t)n * DIM, lane);
      gemm_acc<DIM>(acc.v, cat16, 512, Whh_r + (size_t)n * DIM, lane);
      int m0 = (lane >> 4) << 3;
#pragma unroll
      for (int g = 0; g < 8; ++g) gates[(m0 + g) * 1024 + n] = acc.f[g];
    }
    __syncthreads();

    // ---- read-LSTM cell (f32), h_r -> cat16[:,0:256] (f16) ----
    for (int i = tid; i < BPG * DIM; i += NTHR) {
      int b = i >> 8, d = i & 255;
      const float* gr = gates + b * 1024;
      float ig = sigmoidf_fast(gr[d]);
      float fg = sigmoidf_fast(gr[256 + d]);
      float gg = tanhf(gr[512 + d]);
      float og = sigmoidf_fast(gr[768 + d]);
      float c  = fg * c_r[i] + ig * gg;
      c_r[i]   = c;
      cat16[b * 512 + d] = (_Float16)(og * tanhf(c));
    }
    __syncthreads();

    // ---- attention scores: z[b][s] = h_r[b] . M[b][s]  (float4 rows) ----
    for (int i = tid; i < BPG * S_LEN; i += NTHR) {
      int b = i / S_LEN, s = i - b * S_LEN;
      const float4*   mrow = (const float4*)(M + (size_t)(b0 + b) * S_LEN * DIM + s * DIM);
      const _Float16* hrow = cat16 + b * 512;
      float dot = 0.0f;
#pragma unroll 4
      for (int d4 = 0; d4 < DIM / 4; ++d4) {
        float4 mv = mrow[d4];
        dot += (float)hrow[4 * d4 + 0] * mv.x + (float)hrow[4 * d4 + 1] * mv.y +
               (float)hrow[4 * d4 + 2] * mv.z + (float)hrow[4 * d4 + 3] * mv.w;
      }
      zbuf[i] = dot;
    }
    __syncthreads();

    // ---- softmax: one wave per batch row, lane-parallel over 96 slots ----
    {
      float* zr = zbuf + wave * S_LEN;          // NW == BPG == 16
      float mx = -3.0e38f;
      for (int s = lane; s < S_LEN; s += 32) mx = fmaxf(mx, zr[s]);
#pragma unroll
      for (int off = 16; off; off >>= 1) mx = fmaxf(mx, __shfl_xor(mx, off, 32));
      float sum = 0.0f;
      for (int s = lane; s < S_LEN; s += 32) {
        float e = __expf(zr[s] - mx);
        zr[s] = e;
        sum += e;
      }
#pragma unroll
      for (int off = 16; off; off >>= 1) sum += __shfl_xor(sum, off, 32);
      float inv = 1.0f / sum;
      for (int s = lane; s < S_LEN; s += 32) zr[s] *= inv;
    }
    __syncthreads();

    // ---- read vector: m_r[b][d] = sum_s z[b][s] * M[b][s][d] -> cat16[:,256:512] ----
    for (int i = tid; i < BPG * DIM; i += NTHR) {
      int b = i >> 8, d = i & 255;
      const float* mcol = M + (size_t)(b0 + b) * S_LEN * DIM + d;
      const float* zr   = zbuf + b * S_LEN;
      float acc = 0.0f;
#pragma unroll 4
      for (int s = 0; s < S_LEN; ++s) acc += zr[s] * mcol[s * DIM];
      cat16[b * 512 + 256 + d] = (_Float16)acc;
    }
    __syncthreads();

    // ---- GEMM2: comp = cat @ Wx^T + bx  (16x512) ----
    for (int tt = wave; tt < 32; tt += NW) {
      int n = tt * 16 + (lane & 15);
      float bias = bx[n];
      V8F acc;
#pragma unroll
      for (int g = 0; g < 8; ++g) acc.f[g] = bias;
      gemm_acc<512>(acc.v, cat16, 512, Wx + (size_t)n * 512, lane);
      int m0 = (lane >> 4) << 3;
#pragma unroll
      for (int g = 0; g < 8; ++g) comp16[(m0 + g) * 512 + n] = (_Float16)acc.f[g];
    }
    __syncthreads();

    // ---- GEMM3: gates = comp @ Wih_w^T + h_w @ Whh_w^T + b_w  (16x1024) ----
    for (int tt = wave; tt < 64; tt += NW) {
      int n = tt * 16 + (lane & 15);
      float bias = b_w[n];
      V8F acc;
#pragma unroll
      for (int g = 0; g < 8; ++g) acc.f[g] = bias;
      gemm_acc<512>(acc.v, comp16, 512, Wih_w + (size_t)n * 512, lane);
      gemm_acc<DIM>(acc.v, hw16,  DIM, Whh_w + (size_t)n * DIM, lane);
      int m0 = (lane >> 4) << 3;
#pragma unroll
      for (int g = 0; g < 8; ++g) gates[(m0 + g) * 1024 + n] = acc.f[g];
    }
    __syncthreads();

    // ---- write-LSTM cell, h_w -> hw16 (f16) ----
    for (int i = tid; i < BPG * DIM; i += NTHR) {
      int b = i >> 8, d = i & 255;
      const float* gr = gates + b * 1024;
      float ig = sigmoidf_fast(gr[d]);
      float fg = sigmoidf_fast(gr[256 + d]);
      float gg = tanhf(gr[512 + d]);
      float og = sigmoidf_fast(gr[768 + d]);
      float c  = fg * c_w[i] + ig * gg;
      c_w[i]   = c;
      hw16[i]  = (_Float16)(og * tanhf(c));
    }
    __syncthreads();

    // ---- memory update: M = M*(1-z) + h_w*z  (float4 vectorized) ----
    const int F4 = BPG * S_LEN * (DIM / 4);
    for (int i = tid; i < F4; i += NTHR) {
      int d4 = i & 63;
      int s  = (i >> 6) % S_LEN;
      int b  = i / (S_LEN * 64);
      float zv = zbuf[b * S_LEN + s];
      float om = 1.0f - zv;
      float4* p = (float4*)(M + (size_t)(b0 + b) * S_LEN * DIM + s * DIM + d4 * 4);
      float4 mv = *p;
      const _Float16* hb = hw16 + b * DIM + d4 * 4;
      mv.x = mv.x * om + (float)hb[0] * zv;
      mv.y = mv.y * om + (float)hb[1] * zv;
      mv.z = mv.z * om + (float)hb[2] * zv;
      mv.w = mv.w * om + (float)hb[3] * zv;
      *p = mv;
    }
    __threadfence();   // make M stores visible to this WG's next-step reads
    __syncthreads();
  }
}

// ------------------------------------------------------------------ entry ---

extern "C" void kernel_launch(void* const* d_in, const int* in_sizes, int n_in,
                              void* d_out, int out_size, void* d_ws, size_t ws_size,
                              hipStream_t stream) {
  (void)in_sizes; (void)n_in; (void)out_size; (void)ws_size;
  const int*   premise = (const int*)d_in[0];
  const float* embed_W = (const float*)d_in[1];
  const float* Wih_r   = (const float*)d_in[2];
  const float* Whh_r   = (const float*)d_in[3];
  const float* b_r     = (const float*)d_in[4];
  const float* Wx      = (const float*)d_in[5];
  const float* bx      = (const float*)d_in[6];
  const float* Wih_w   = (const float*)d_in[7];
  const float* Whh_w   = (const float*)d_in[8];
  const float* b_w     = (const float*)d_in[9];
  float* M = (float*)d_out;

  // f16 weight copies in workspace (3 MB total)
  _Float16* ws      = (_Float16*)d_ws;
  _Float16* Wih_r16 = ws;             // 1024*256
  _Float16* Whh_r16 = ws + 262144;    // 1024*256
  _Float16* Wx16    = ws + 524288;    // 512*512
  _Float16* Wih_w16 = ws + 786432;    // 1024*512
  _Float16* Whh_w16 = ws + 1310720;   // 1024*256

  NSE_f32_to_f16<<<(1024*256 + 255) / 256, 256, 0, stream>>>(Wih_r, Wih_r16, 1024*256);
  NSE_f32_to_f16<<<(1024*256 + 255) / 256, 256, 0, stream>>>(Whh_r, Whh_r16, 1024*256);
  NSE_f32_to_f16<<<(512*512  + 255) / 256, 256, 0, stream>>>(Wx,    Wx16,    512*512);
  NSE_f32_to_f16<<<(1024*512 + 255) / 256, 256, 0, stream>>>(Wih_w, Wih_w16, 1024*512);
  NSE_f32_to_f16<<<(1024*256 + 255) / 256, 256, 0, stream>>>(Whh_w, Whh_w16, 1024*256);

  NSE_init_M<<<(BATCH * S_LEN * DIM + 255) / 256, 256, 0, stream>>>(premise, embed_W, M);

  // 4 workgroups x 16 batch rows, 142 KB dynamic LDS each
  NSE_scan<<<BATCH / BPG, NTHR, 145408, stream>>>(
      premise, embed_W, Wih_r16, Whh_r16, b_r, Wx16, bx, Wih_w16, Whh_w16, b_w, M);
}
